// _Head_33011118637665
// MI455X (gfx1250) — matrix operand order repaired and checked by reading, not compiled, not run on hardware
//
#include <hip/hip_runtime.h>
#include <hip/hip_bf16.h>

// MI455X / gfx1250 fused attention head (QKV proj + causal flash attention).
// Matmuls via v_wmma_f32_16x16x32_f16; operand staging via the CDNA5 async
// global->LDS path (global_load_async_to_lds_b128, ASYNCcnt) double-buffered.

typedef __attribute__((ext_vector_type(16))) _Float16 v16h;
typedef __attribute__((ext_vector_type(8)))  float    v8f;

#define WMMA_F16F32(A, B, C) \
    __builtin_amdgcn_wmma_f32_16x16x32_f16(false, (A), false, (B), (short)0, (C), false, false)

#define HEAD_DIM 64
#define EMB_DIM  1024
#define BATCH    4
#define SEQ      4096
#define ROWS     (BATCH * SEQ)      // 16384
#define WPB      4                  // waves per block (128 threads)

// ---- CDNA5 async global->LDS copy (tracked by ASYNCcnt) --------------------
// NOTE: lds_base is passed (and printed in an asm comment) purely so the
// shared-memory object ESCAPES: otherwise LLVM proves the asm cannot write
// the LDS array and folds all its loads to undef (observed in round 2).
__device__ __forceinline__ void async_ld_b128(unsigned lds_byte_off, const void* gaddr,
                                              const void* lds_base) {
    asm volatile("global_load_async_to_lds_b128 %0, %1, off ;; lds base %2"
                 :: "v"(lds_byte_off), "v"(gaddr), "v"(lds_base)
                 : "memory");
}
#define WAIT_ASYNC_LE(lit) asm volatile("s_wait_asynccnt " lit ::: "memory")

// ---------------------------------------------------------------------------
// Kernel 1: transpose + convert weights  W[E,D] f32 -> Wt[D,E] f16 (x scale).
// ---------------------------------------------------------------------------
__global__ void __launch_bounds__(256)
head_wconv_kernel(const float* __restrict__ W, _Float16* __restrict__ Wt, float scale) {
    int i = blockIdx.x * blockDim.x + threadIdx.x;       // 0 .. E*D-1
    if (i < EMB_DIM * HEAD_DIM) {
        int e = i >> 6;          // / HEAD_DIM
        int d = i & 63;
        Wt[d * EMB_DIM + e] = (_Float16)(W[i] * scale);
    }
}

// ---------------------------------------------------------------------------
// Kernel 2: QKV projection. One wave computes a 16x64 tile of Q, K and V.
// Weight K-slices (3 mats x 64 rows x 32 halves = 12KB) are staged into a
// 2x12KB LDS double buffer with async loads shared by all 4 waves.
//   Q -> Qh [ROWS,64] f16 ; K -> Kh [ROWS,64] f16 ; V -> Vt [B,64,SEQ] f16
// ---------------------------------------------------------------------------
__global__ void __launch_bounds__(WPB * 32)
head_qkv_kernel(const float* __restrict__ x,
                const _Float16* __restrict__ Wall,     // Wqt|Wkt|Wvt contiguous [3][64][1024]
                _Float16* __restrict__ Qh,
                _Float16* __restrict__ Kh,
                _Float16* __restrict__ Vt) {
    __shared__ _Float16 wbuf[2 * 3 * 64 * 32];          // 2 x 12KB, LDS offset 0

    const int t    = threadIdx.x;
    const int lane = t & 31;
    const int wave = t >> 5;
    const int g    = lane >> 4;
    const int ln   = lane & 15;
    const int tile = blockIdx.x * WPB + wave;
    const int row0 = tile * 16;

    // Per-thread async chunk mapping: 12KB = 768 x 16B, 6 chunks/thread.
    // chunk c -> mat m = c>>8, row d = (c&255)>>2, part p = c&3
    // global half offset = m*65536 + d*1024 + p*8 (+k0) ; LDS byte off = c*16
    unsigned goff[6], loff[6];
#pragma unroll
    for (int q = 0; q < 6; ++q) {
        const int c = q * 128 + t;
        const int m = c >> 8, r = c & 255, d = r >> 2, p = r & 3;
        goff[q] = (unsigned)(m * 65536 + d * 1024 + p * 8);
        loff[q] = (unsigned)(c * 16);
    }

    // prologue: stage k0 = 0 into buffer 0
#pragma unroll
    for (int q = 0; q < 6; ++q)
        async_ld_b128(loff[q], (const char*)Wall + (size_t)goff[q] * 2, wbuf);

    v8f acc[12];                                        // [0..3]=Q [4..7]=K [8..11]=V
#pragma unroll
    for (int i = 0; i < 12; ++i)
#pragma unroll
        for (int r = 0; r < 8; ++r) acc[i][r] = 0.f;

    const float* xrow = x + (size_t)(row0 + ln) * EMB_DIM;

    int cur = 0;
    for (int k0 = 0; k0 < EMB_DIM; k0 += 32, cur ^= 1) {
        if (k0 + 32 < EMB_DIM) {
#pragma unroll
            for (int q = 0; q < 6; ++q)
                async_ld_b128((unsigned)((cur ^ 1) * 12288) + loff[q],
                              (const char*)Wall + (size_t)(goff[q] + k0 + 32) * 2, wbuf);
            WAIT_ASYNC_LE("0x6");                       // current buffer complete
        } else {
            WAIT_ASYNC_LE("0x0");
        }
        __syncthreads();

        // A fragment (x rows, f32->f16): contiguous k = 16*g + i layout.
        v16h a;
#pragma unroll
        for (int i = 0; i < 8; ++i) {
            a[i]     = (_Float16)xrow[k0 + g * 16 + i];
            a[8 + i] = (_Float16)xrow[k0 + g * 16 + 8 + i];
        }

        const int bufh = cur * 6144;                    // halves per 12KB buffer
#pragma unroll
        for (int m = 0; m < 3; ++m)
#pragma unroll
            for (int nt = 0; nt < 4; ++nt) {
                const int o = bufh + m * 2048 + (nt * 16 + ln) * 32 + g * 16;
                v16h bf;
#pragma unroll
                for (int i = 0; i < 16; ++i) bf[i] = wbuf[o + i];   // 2x ds_load_b128
                acc[m * 4 + nt] = WMMA_F16F32(a, bf, acc[m * 4 + nt]);
            }
        __syncthreads();
    }

    // C/D layout: VGPR r <-> row (r + 8*g), lane%16 <-> column.
#pragma unroll
    for (int nt = 0; nt < 4; ++nt)
#pragma unroll
        for (int r = 0; r < 8; ++r) {
            const int row = row0 + r + 8 * g;
            const int col = nt * 16 + ln;
            Qh[(size_t)row * HEAD_DIM + col] = (_Float16)acc[nt][r];
            Kh[(size_t)row * HEAD_DIM + col] = (_Float16)acc[4 + nt][r];
            const int b = row >> 12;
            const int s = row & (SEQ - 1);
            Vt[((size_t)(b * HEAD_DIM + col)) * SEQ + s] = (_Float16)acc[8 + nt][r];
        }
}

// ---------------------------------------------------------------------------
// Kernel 3: causal flash attention. 4 waves/block share one batch and an
// overlapping key range, so each 32-key K chunk (4KB, [32][64]) and V chunk
// (4KB, [64][32]) is staged cooperatively into a 2x8KB async double buffer.
// smem halves: [0..8191] KV buffers, [8192..10239] P-transpose tiles.
// ---------------------------------------------------------------------------
__global__ void __launch_bounds__(WPB * 32)
head_attn_kernel(const _Float16* __restrict__ Qh,
                 const _Float16* __restrict__ Kh,
                 const _Float16* __restrict__ Vt,
                 float* __restrict__ out) {
    __shared__ _Float16 smem[10240];                    // 20KB, LDS offset 0

    const int t    = threadIdx.x;
    const int lane = t & 31;
    const int wave = t >> 5;
    const int g    = lane >> 4;
    const int ln   = lane & 15;
    const int tile = blockIdx.x * WPB + wave;
    const int b    = tile >> 8;
    const int q0   = (tile & 255) << 4;
    const size_t bbase = (size_t)b * SEQ;
    const int q0max = ((blockIdx.x * WPB + (WPB - 1)) & 255) << 4;  // block-uniform
    const int jend  = q0max + 16;

    // Per-thread staging mapping (k0-independent global half offsets).
    // K chunks c = t, t+128: row jr = c>>3, part p = c&7
    // V chunks c = t, t+128: row d  = c>>2, part p = c&3
    size_t kgo[2], vgo[2];
    unsigned klo[2], vlo[2];
#pragma unroll
    for (int q = 0; q < 2; ++q) {
        const int c = q * 128 + t;
        kgo[q] = (bbase + (c >> 3)) * HEAD_DIM + (c & 7) * 8;
        vgo[q] = ((size_t)(b * HEAD_DIM + (c >> 2))) * SEQ + (c & 3) * 8;
        klo[q] = (unsigned)(c * 16);                    // K region byte offset
        vlo[q] = (unsigned)(4096 + c * 16);             // V region byte offset
    }

    // Q as two A-fragments (d = 0..31 / 32..63); 1/sqrt(d) folded into Wq.
    const _Float16* qrow = Qh + (bbase + q0 + ln) * HEAD_DIM;
    v16h qa0, qa1;
#pragma unroll
    for (int i = 0; i < 8; ++i) {
        qa0[i]     = qrow[g * 16 + i];
        qa0[8 + i] = qrow[g * 16 + 8 + i];
        qa1[i]     = qrow[32 + g * 16 + i];
        qa1[8 + i] = qrow[40 + g * 16 + i];
    }

    v8f o0, o1, o2, o3;
    float mrow[8], lrow[8];
#pragma unroll
    for (int r = 0; r < 8; ++r) {
        o0[r] = 0.f; o1[r] = 0.f; o2[r] = 0.f; o3[r] = 0.f;
        mrow[r] = -1e30f; lrow[r] = 0.f;
    }

    const int pbase = 8192 + wave * 512;                // P tile (halves)

    // prologue: stage chunk j0 = 0 into buffer 0
#pragma unroll
    for (int q = 0; q < 2; ++q) {
        async_ld_b128(klo[q], (const char*)Kh + kgo[q] * 2, smem);
        async_ld_b128(vlo[q], (const char*)Vt + vgo[q] * 2, smem);
    }

    int cur = 0;
    for (int j0 = 0; j0 < jend; j0 += 32, cur ^= 1) {
        if (j0 + 32 < jend) {                           // block-uniform branch
            const unsigned nb = (unsigned)((cur ^ 1) * 8192);
#pragma unroll
            for (int q = 0; q < 2; ++q) {
                async_ld_b128(nb + klo[q],
                              (const char*)Kh + (kgo[q] + (size_t)(j0 + 32) * HEAD_DIM) * 2, smem);
                async_ld_b128(nb + vlo[q],
                              (const char*)Vt + (vgo[q] + (size_t)(j0 + 32)) * 2, smem);
            }
            WAIT_ASYNC_LE("0x4");
        } else {
            WAIT_ASYNC_LE("0x0");
        }
        __syncthreads();

        if (j0 < q0 + 16) {                             // this wave's causal range
            const int bufh = cur * 4096;                // halves
            // ---- scores: two 16-key subtiles, K^T B-frags from LDS ----
            v8f s0, s1;
#pragma unroll
            for (int r = 0; r < 8; ++r) { s0[r] = 0.f; s1[r] = 0.f; }
            {
                const int k0o = bufh + (ln)      * 64 + g * 16;   // subtile 0 row
                const int k1o = bufh + (16 + ln) * 64 + g * 16;   // subtile 1 row
                v16h kb00, kb01, kb10, kb11;
#pragma unroll
                for (int i = 0; i < 16; ++i) {
                    kb00[i] = smem[k0o + i];
                    kb01[i] = smem[k0o + 32 + i];
                    kb10[i] = smem[k1o + i];
                    kb11[i] = smem[k1o + 32 + i];
                }
                s0 = WMMA_F16F32(qa0, kb00, s0);
                s0 = WMMA_F16F32(qa1, kb01, s0);
                s1 = WMMA_F16F32(qa0, kb10, s1);
                s1 = WMMA_F16F32(qa1, kb11, s1);
            }

            // ---- causal mask + online softmax (row = q0+r+8g, col = j0+...) ----
#pragma unroll
            for (int r = 0; r < 8; ++r) {
                const int mg = q0 + r + 8 * g;
                float c0 = (j0 + ln      <= mg) ? s0[r] : -1e30f;
                float c1 = (j0 + 16 + ln <= mg) ? s1[r] : -1e30f;
                float rmax = fmaxf(c0, c1);
                rmax = fmaxf(rmax, __shfl_xor(rmax, 1, 32));
                rmax = fmaxf(rmax, __shfl_xor(rmax, 2, 32));
                rmax = fmaxf(rmax, __shfl_xor(rmax, 4, 32));
                rmax = fmaxf(rmax, __shfl_xor(rmax, 8, 32));
                const float mnew  = fmaxf(mrow[r], rmax);
                const float scale = __expf(mrow[r] - mnew);
                mrow[r] = mnew;
                const float p0 = __expf(c0 - mnew);
                const float p1 = __expf(c1 - mnew);
                lrow[r] = lrow[r] * scale + p0 + p1;    // per-lane partial row sum
                o0[r] *= scale; o1[r] *= scale; o2[r] *= scale; o3[r] *= scale;
                smem[pbase + (r + 8 * g) * 32 + ln]      = (_Float16)p0;
                smem[pbase + (r + 8 * g) * 32 + 16 + ln] = (_Float16)p1;
            }

            // ---- reload P as 16x32 A-fragment (wave-private, DS in-order) ----
            v16h pa;
            {
                const int pr = pbase + ln * 32 + g * 16;
#pragma unroll
                for (int i = 0; i < 8; ++i) { pa[i] = smem[pr + i]; pa[8 + i] = smem[pr + 8 + i]; }
            }

            // ---- O += P x V : V B-frags contiguous from LDS [64][32] ----
            {
                const int vb = bufh + 2048 + ln * 32 + g * 16;
                v16h vf0, vf1, vf2, vf3;
#pragma unroll
                for (int i = 0; i < 16; ++i) {
                    vf0[i] = smem[vb + i];
                    vf1[i] = smem[vb + 16 * 32 + i];
                    vf2[i] = smem[vb + 32 * 32 + i];
                    vf3[i] = smem[vb + 48 * 32 + i];
                }
                o0 = WMMA_F16F32(pa, vf0, o0);
                o1 = WMMA_F16F32(pa, vf1, o1);
                o2 = WMMA_F16F32(pa, vf2, o2);
                o3 = WMMA_F16F32(pa, vf3, o3);
            }
        }
        __syncthreads();
    }

    // ---- finalize: reduce row sums, normalize, store fp32 ----
#pragma unroll
    for (int r = 0; r < 8; ++r) {
        float l = lrow[r];
        l += __shfl_xor(l, 1, 32);
        l += __shfl_xor(l, 2, 32);
        l += __shfl_xor(l, 4, 32);
        l += __shfl_xor(l, 8, 32);
        const float inv = 1.0f / l;
        const size_t obase = (bbase + q0 + r + 8 * g) * HEAD_DIM;
        out[obase + ln]      = o0[r] * inv;
        out[obase + 16 + ln] = o1[r] * inv;
        out[obase + 32 + ln] = o2[r] * inv;
        out[obase + 48 + ln] = o3[r] * inv;
    }
}

// ---------------------------------------------------------------------------
// Workspace: Wqt|Wkt|Wvt (3 x 128KB, contiguous), Qh (2MB), Kh (2MB), Vt (2MB)
// ---------------------------------------------------------------------------
extern "C" void kernel_launch(void* const* d_in, const int* in_sizes, int n_in,
                              void* d_out, int out_size, void* d_ws, size_t ws_size,
                              hipStream_t stream) {
    const float* x  = (const float*)d_in[0];
    const float* Wq = (const float*)d_in[1];
    const float* Wk = (const float*)d_in[2];
    const float* Wv = (const float*)d_in[3];
    float* out = (float*)d_out;

    char* ws = (char*)d_ws;
    const size_t WSZ = (size_t)EMB_DIM * HEAD_DIM * sizeof(_Float16);   // 128 KB
    const size_t QSZ = (size_t)ROWS * HEAD_DIM * sizeof(_Float16);      // 2 MB
    _Float16* Wqt = (_Float16*)(ws);                    // contiguous Wall
    _Float16* Wkt = (_Float16*)(ws + WSZ);
    _Float16* Wvt = (_Float16*)(ws + 2 * WSZ);
    _Float16* Qh  = (_Float16*)(ws + 3 * WSZ);
    _Float16* Kh  = (_Float16*)(ws + 3 * WSZ + QSZ);
    _Float16* Vt  = (_Float16*)(ws + 3 * WSZ + 2 * QSZ);

    head_wconv_kernel<<<(EMB_DIM * HEAD_DIM) / 256, 256, 0, stream>>>(Wq, Wqt, 0.125f);
    head_wconv_kernel<<<(EMB_DIM * HEAD_DIM) / 256, 256, 0, stream>>>(Wk, Wkt, 1.0f);
    head_wconv_kernel<<<(EMB_DIM * HEAD_DIM) / 256, 256, 0, stream>>>(Wv, Wvt, 1.0f);

    head_qkv_kernel<<<ROWS / 16 / WPB, WPB * 32, 0, stream>>>(x, Wqt, Qh, Kh, Vt);

    head_attn_kernel<<<ROWS / 16 / WPB, WPB * 32, 0, stream>>>(Qh, Kh, Vt, out);
}
